// DotAttention_1400159339109
// MI455X (gfx1250) — matrix operand order repaired and checked by reading, not compiled, not run on hardware
//
#include <hip/hip_runtime.h>

typedef __attribute__((ext_vector_type(16))) _Float16 v16h;
typedef __attribute__((ext_vector_type(8)))  _Float16 v8h;
typedef __attribute__((ext_vector_type(4)))  _Float16 v4h;
typedef __attribute__((ext_vector_type(8)))  float    v8f;
typedef __attribute__((ext_vector_type(4)))  float    v4f;

#define B_DIM    32
#define TD       512
#define TE       512
#define D_DIM    512
#define MT       32    // decoder rows per workgroup
#define EB       64    // encoder rows staged per phase-1 block
#define DB       64    // context columns per phase-3 block
#define NTHREADS 128   // 4 waves (wave32)

// LDS carve (160 KB total, dynamic):
//   [0      , 65536)  S      : 32x512 f32 scores
//   [65536  , 131072) bufA   : 64x512 f16  (enc block / encT block; rowstats alias)
//   [131072 , 163840) bufB   : 32x512 f16  (dec tile, then P tile)
#define SMEM_BYTES 163840

// ---- WMMA fragment loads from row-major f16 LDS arrays (row stride 512) ----
// A (16x32 f16, MxK): lane L holds row M=L&15; contiguous K-run [h*16,+16), h=L>>4.
__device__ __forceinline__ v16h frag_a(const _Float16* base, int row0, int k0, int lane) {
  const int r = row0 + (lane & 15);
  const int h = lane >> 4;
  const _Float16* p = base + r * 512 + k0 + h * 16;
  return *(const v16h*)(p);
}
// B (32x16 f16, KxN): lane L holds column N=L&15; contiguous K-run [h*16,+16), h=L>>4.
__device__ __forceinline__ v16h frag_b(const _Float16* base, int col0, int k0, int lane) {
  const int n = col0 + (lane & 15);
  const int h = lane >> 4;
  return *(const v16h*)(base + n * 512 + k0 + h * 16);
}
#define WMMA(a, b, c) \
  __builtin_amdgcn_wmma_f32_16x16x32_f16(false, (a), false, (b), (short)0, (c), false, false)

__global__ void DotAttention_1400159339109_kernel(const float* __restrict__ enc,
                                                  const float* __restrict__ dec,
                                                  float* __restrict__ ctx,
                                                  float* __restrict__ attn) {
  extern __shared__ char smem[];
  float*    S        = (float*)smem;                       // 32x512 f32
  _Float16* bufA     = (_Float16*)(smem + 65536);          // 64x512 f16
  _Float16* bufB     = (_Float16*)(smem + 131072);         // 32x512 f16
  float*    rowstats = (float*)(smem + 65536);             // alias bufA (phase 2 only)

  const int tid  = threadIdx.x;
  const int lane = tid & 31;
  const int w    = tid >> 5;            // wave id 0..3
  const int b    = blockIdx.x >> 4;     // batch
  const int m0   = (blockIdx.x & 15) * MT;

  const float* encB = enc + (size_t)b * TE * D_DIM;
  const float* decB = dec + ((size_t)b * TD + m0) * D_DIM;

  // Per-wave tile assignment: 2 m-tiles x 4 n-tiles = 8 tiles, 2 per wave.
  const int mt  = w & 1;                // m-tile (16 rows)
  const int ntb = (w >> 1) * 2;         // first of two n-tiles (16 cols each)
  const int hl  = lane >> 4;            // C/D layout: VGPR i -> row i + 8*hl
  const int nl  = lane & 15;            // C/D layout: col = lane&15

  // ---------- stage dec tile (32x512 fp32 -> f16) ----------
  for (int i4 = tid; i4 < MT * (D_DIM / 4); i4 += NTHREADS) {
    const int r = i4 >> 7, q = i4 & 127;
    v4f v = *(const v4f*)(decB + r * D_DIM + q * 4);
    v4h hv = { (_Float16)v.x, (_Float16)v.y, (_Float16)v.z, (_Float16)v.w };
    *(v4h*)(bufB + r * 512 + q * 4) = hv;
  }

  // ================= Phase 1: S = dec @ enc^T =================
  for (int eb = 0; eb < TE / EB; ++eb) {
    const int e0 = eb * EB;
    __syncthreads();  // prior block's B-frag reads done; dec staging visible (eb==0)
    for (int i4 = tid; i4 < EB * (D_DIM / 4); i4 += NTHREADS) {
      const int el = i4 >> 7, q = i4 & 127;
      v4f v = *(const v4f*)(encB + (e0 + el) * D_DIM + q * 4);
      v4h hv = { (_Float16)v.x, (_Float16)v.y, (_Float16)v.z, (_Float16)v.w };
      *(v4h*)(bufA + el * 512 + q * 4) = hv;
      if (eb + 1 < TE / EB)
        __builtin_prefetch(encB + (e0 + EB + el) * D_DIM + q * 4, 0, 1);
    }
    __syncthreads();

    v8f acc0 = {}, acc1 = {};
#pragma unroll 4
    for (int k0 = 0; k0 < D_DIM; k0 += 32) {
      v16h a  = frag_a(bufB, mt * 16, k0, lane);
      v16h b0 = frag_b(bufA, ntb * 16, k0, lane);
      v16h b1 = frag_b(bufA, ntb * 16 + 16, k0, lane);
      acc0 = WMMA(a, b0, acc0);
      acc1 = WMMA(a, b1, acc1);
    }
#pragma unroll
    for (int i = 0; i < 8; ++i) {
      const int r = mt * 16 + i + 8 * hl;
      S[r * 512 + e0 + ntb * 16 + nl]      = acc0[i];
      S[r * 512 + e0 + ntb * 16 + 16 + nl] = acc1[i];
    }
  }
  __syncthreads();

  // ================= Phase 2a: per-row max & 1/sum =================
  {
    const int row = w * 8 + (lane >> 2);
    const int q   = lane & 3;
    float m = -3.4e38f;
    for (int c = q; c < TE; c += 4) m = fmaxf(m, S[row * 512 + c]);
    m = fmaxf(m, __shfl_xor(m, 1, 32));
    m = fmaxf(m, __shfl_xor(m, 2, 32));
    float s = 0.f;
    for (int c = q; c < TE; c += 4) s += __expf(S[row * 512 + c] - m);
    s += __shfl_xor(s, 1, 32);
    s += __shfl_xor(s, 2, 32);
    if (q == 0) { rowstats[row * 2] = m; rowstats[row * 2 + 1] = 1.0f / s; }
  }
  __syncthreads();

  // ===== Phase 2b: normalize, write attention (coalesced) + P f16 to LDS =====
  {
    float* attnB = attn + ((size_t)b * TD + m0) * TE;
    for (int idx = tid; idx < MT * TE; idx += NTHREADS) {
      const int row = idx >> 9;
      const float p = __expf(S[idx] - rowstats[row * 2]) * rowstats[row * 2 + 1];
      attnB[idx] = p;               // consecutive tid -> consecutive t_enc
      bufB[idx]  = (_Float16)p;     // P tile (overwrites dec16, no longer needed)
    }
  }

  // ================= Phase 3: context = P @ enc =================
  for (int db = 0; db < D_DIM / DB; ++db) {
    const int d0 = db * DB;
    __syncthreads();  // prior block's encT reads done; rowstats dead after 2b
    // stage encT f16: bufA[d_local][e] = enc[e][d0+d_local]
    for (int i = tid; i < TE * (DB / 4); i += NTHREADS) {
      const int e = i >> 4, q = i & 15;
      v4f v = *(const v4f*)(encB + e * D_DIM + d0 + q * 4);
      bufA[(q * 4 + 0) * 512 + e] = (_Float16)v.x;
      bufA[(q * 4 + 1) * 512 + e] = (_Float16)v.y;
      bufA[(q * 4 + 2) * 512 + e] = (_Float16)v.z;
      bufA[(q * 4 + 3) * 512 + e] = (_Float16)v.w;
      if (db + 1 < D_DIM / DB)
        __builtin_prefetch(encB + e * D_DIM + d0 + DB + q * 4, 0, 1);
    }
    __syncthreads();

    v8f acc0 = {}, acc1 = {};
#pragma unroll 4
    for (int k0 = 0; k0 < TE; k0 += 32) {  // contract over encoder positions
      v16h a  = frag_a(bufB, mt * 16, k0, lane);           // P tile
      v16h b0 = frag_b(bufA, ntb * 16, k0, lane);          // encT tile
      v16h b1 = frag_b(bufA, ntb * 16 + 16, k0, lane);
      acc0 = WMMA(a, b0, acc0);
      acc1 = WMMA(a, b1, acc1);
    }
    float* ctxB = ctx + ((size_t)b * TD + m0) * D_DIM;
#pragma unroll
    for (int i = 0; i < 8; ++i) {
      const int r = mt * 16 + i + 8 * hl;
      ctxB[r * D_DIM + d0 + ntb * 16 + nl]      = acc0[i];
      ctxB[r * D_DIM + d0 + ntb * 16 + 16 + nl] = acc1[i];
    }
  }
}

extern "C" void kernel_launch(void* const* d_in, const int* in_sizes, int n_in,
                              void* d_out, int out_size, void* d_ws, size_t ws_size,
                              hipStream_t stream) {
  const float* enc = (const float*)d_in[0];   // (B, Te, D) fp32
  const float* dec = (const float*)d_in[1];   // (B, Td, D) fp32
  float* ctx  = (float*)d_out;                                // (B, Td, D)
  float* attn = (float*)d_out + (size_t)B_DIM * TD * D_DIM;   // (B, Td, Te)

  dim3 grid(B_DIM * (TD / MT));   // 512 workgroups
  dim3 block(NTHREADS);           // 4 waves of 32
  DotAttention_1400159339109_kernel<<<grid, block, SMEM_BYTES, stream>>>(enc, dec, ctx, attn);
}
